// multi_head_attention_52467320488564
// MI455X (gfx1250) — matrix operand order repaired and checked by reading, not compiled, not run on hardware
//
#include <hip/hip_runtime.h>
#include <hip/hip_bf16.h>

typedef __attribute__((ext_vector_type(16))) __bf16 v16bf;
typedef __attribute__((ext_vector_type(8)))  __bf16 v8bf;
typedef __attribute__((ext_vector_type(8)))  float  v8f;
typedef __attribute__((ext_vector_type(4)))  unsigned int v4u;
typedef __attribute__((ext_vector_type(8)))  int v8i_t;
typedef __attribute__((ext_vector_type(4)))  int v4i_t;

static constexpr int Bsz = 2, Ssz = 2048, Dsz = 1024, Hh = 16, DHd = 64;

// ---- gfx1250 Tensor Data Mover path (6-arg builtin on this toolchain:
//      (uint32x4 g0, int32x8 g1, int32x4 g2, int32x4 g3, int32x8, i32 cpol));
//      guarded, with synchronous LDS staging as fallback ----
#if defined(__has_builtin)
#if __has_builtin(__builtin_amdgcn_tensor_load_to_lds) && \
    __has_builtin(__builtin_amdgcn_s_wait_tensorcnt)
#define HAVE_TDM 1
#endif
#endif

#ifdef HAVE_TDM
// 2D tile DMA: tileH rows of tileW bf16 elements (row stride = strideElems),
// packed contiguously into LDS at byte offset ldsOff. D# per CDNA5 ISA ch.8.
__device__ __forceinline__ void tdm_load_2d(unsigned ldsOff, const void* g,
                                            unsigned tileW, unsigned tileH,
                                            unsigned strideElems) {
    unsigned long long ga = (unsigned long long)g;
    v4u g0 = { 1u,                                        // count=1, is_restore=0
               ldsOff,                                    // lds_addr [63:32]
               (unsigned)(ga & 0xFFFFFFFFu),              // global_addr lo
               (unsigned)((ga >> 32) & 0x1FFFFFFu) | (2u << 30) };  // addr hi | type=2
    v8i_t g1 = {
        (int)(1u << 16),                                            // data_size=1 (2B)
        (int)((tileW & 0xFFFFu) << 16),                             // tensor_dim0 lo16
        (int)(((tileW >> 16) & 0xFFFFu) | ((tileH & 0xFFFFu) << 16)), // dim0 hi | dim1 lo
        (int)(((tileH >> 16) & 0xFFFFu) | ((tileW & 0xFFFFu) << 16)), // dim1 hi | tile_dim0
        (int)(tileH & 0xFFFFu),                                     // tile_dim1 (tile_dim2=0)
        (int)strideElems,                                           // tensor_dim0_stride lo32
        0, 0 };
    v4i_t z4 = { 0, 0, 0, 0 };
    v8i_t z8 = { 0, 0, 0, 0, 0, 0, 0, 0 };
    __builtin_amdgcn_tensor_load_to_lds(g0, g1, z4, z4, z8, 0);
}
#endif

// ---------------- WMMA fragment helpers (wave32, gfx1250 layouts) ----------------

// A fragment (16x32 bf16, MxK): lane%16 = M row, lane/16 selects kbase {0,8};
// elements 0..15 -> K=2*kbase+0..15 (16 contiguous K per lane).
__device__ __forceinline__ v16bf load_a_bf(const __bf16* __restrict__ row, int kbase) {
    const __bf16* p = row + 2 * kbase;
    v8bf lo = *(const v8bf*)(p);
    v8bf hi = *(const v8bf*)(p + 8);
    v16bf r;
#pragma unroll
    for (int i = 0; i < 8; ++i) { r[i] = lo[i]; r[i + 8] = hi[i]; }
    return r;
}

// B fragment (32x16 bf16, KxN) from B^T rows: lane%16 = N, lane/16 selects khalf
// {0,16}; 16 contiguous K values per lane.
__device__ __forceinline__ v16bf load_b_bf(const __bf16* __restrict__ rowT, int khalf) {
    return *(const v16bf*)(rowT + khalf);
}

__device__ __forceinline__ v8f wmma_bf16(v16bf a, v16bf b, v8f c) {
    return __builtin_amdgcn_wmma_f32_16x16x32_bf16(
        false, a, false, b, (short)0, c, false, false);
}

// ---------------- Prep kernels ----------------

__global__ void cvt_f32_bf16(const float* __restrict__ src, __bf16* __restrict__ dst, int n) {
    int i = blockIdx.x * blockDim.x + threadIdx.x;
    if (i < n) dst[i] = (__bf16)src[i];
}

__global__ void transpose_cvt(const float* __restrict__ W, __bf16* __restrict__ WT, int Dm) {
    int i = blockIdx.x * blockDim.x + threadIdx.x;
    if (i < Dm * Dm) {
        int k = i / Dm, n = i % Dm;
        WT[(long)n * Dm + k] = (__bf16)W[i];
    }
}

// ---------------- WMMA GEMM with TDM-staged, double-buffered B tiles ----------------
// Block = 256 threads (8 waves) computes a 128(M) x 64(N) tile; the 64x32 bf16
// B-tile per K-step is DMA'd to LDS once (TDM) and shared by all 8 waves.
// mode 0: bf16 [B,H,S,DH] (q, scaled)   mode 1: bf16 [B,H,S,DH] (k)
// mode 2: bf16 [B,H,DH,S] (v^T)         mode 3: fp32 row-major [M,N]
__global__ void gemm_wmma(const __bf16* __restrict__ A, int lda,
                          const __bf16* __restrict__ BT, int ldb,
                          const float* __restrict__ bias,
                          void* __restrict__ out,
                          int M, int N, int K, int mode, float scale) {
    // Only LDS object in this kernel -> LDS byte offset 0 (needed by the D#).
    __shared__ __attribute__((aligned(128))) __bf16 bstage[2][64 * 32];  // 2 x 4 KB

    const int lane = threadIdx.x & 31;
    const int w    = threadIdx.x >> 5;
    const int nstrips = N >> 6;
    const int mblock  = (blockIdx.x / nstrips) << 7;   // 128 rows / block
    const int ntile0  = (blockIdx.x % nstrips) << 6;
    const int mtile   = mblock + (w << 4);
    if (mtile >= M) return;

    const int lm     = lane & 15;
    const int kbaseA = (lane >> 4) << 3;
    const int khalfB = (lane >> 4) << 4;

#ifndef HAVE_TDM
    // fallback staging: each thread copies 16B; 256 threads cover 64x32 bf16
    const int cn = threadIdx.x >> 2;
    const int ck = (threadIdx.x & 3) << 3;
    const __bf16* bsrc = BT + (long)(ntile0 + cn) * ldb + ck;
#endif

    v8f acc[4] = {};
    const __bf16* arow = A + (long)(mtile + lm) * lda;

    // prime buffer 0
#ifdef HAVE_TDM
    if (w == 0)
        tdm_load_2d(0u, BT + (long)ntile0 * ldb, 32u, 64u, (unsigned)ldb);
#else
    *(v8bf*)&bstage[0][cn * 32 + ck] = *(const v8bf*)(bsrc);
#endif

    int buf = 0;
    for (int k0 = 0; k0 < K; k0 += 32) {
        const bool hasNext = (k0 + 32) < K;
#ifdef HAVE_TDM
        if (w == 0) {
            if (hasNext) {
                tdm_load_2d((unsigned)((buf ^ 1) * 4096),
                            BT + (long)ntile0 * ldb + k0 + 32, 32u, 64u, (unsigned)ldb);
                __builtin_amdgcn_s_wait_tensorcnt(1);   // current buffer landed
            } else {
                __builtin_amdgcn_s_wait_tensorcnt(0);
            }
        }
#endif
        __syncthreads();
        __builtin_prefetch(arow + k0 + 128, 0, 1);
        v16bf a = load_a_bf(arow + k0, kbaseA);
#pragma unroll
        for (int t = 0; t < 4; ++t) {
            v16bf b = *(const v16bf*)(&bstage[buf][(t * 16 + lm) * 32 + khalfB]);
            acc[t] = wmma_bf16(a, b, acc[t]);
        }
        __syncthreads();
#ifndef HAVE_TDM
        if (hasNext)
            *(v8bf*)&bstage[buf ^ 1][cn * 32 + ck] = *(const v8bf*)(bsrc + k0 + 32);
#endif
        buf ^= 1;
    }

    const int mrow0 = mtile + ((lane >> 4) << 3);
#pragma unroll
    for (int t = 0; t < 4; ++t) {
        const int n   = ntile0 + t * 16 + lm;
        const float bb = bias ? bias[n] : 0.0f;
#pragma unroll
        for (int j = 0; j < 8; ++j) {
            const int m = mrow0 + j;
            const float vout = (acc[t][j] + bb) * scale;
            if (mode == 3) {
                ((float*)out)[(long)m * N + n] = vout;
            } else {
                const int b  = m / Ssz, s  = m % Ssz;
                const int h  = n / DHd, dh = n % DHd;
                long idx;
                if (mode == 2) idx = (((long)(b * Hh + h)) * DHd + dh) * Ssz + s;
                else           idx = (((long)(b * Hh + h)) * Ssz + s) * DHd + dh;
                ((__bf16*)out)[idx] = (__bf16)vout;
            }
        }
    }
}

// ---------------- Fused attention core: scores -> mask -> softmax -> PV ----------------
// One block (8 waves) per (b,h, 16 q-rows). The 16x2048 fp32 score strip lives
// entirely in (dynamic) LDS: soft_mat hits HBM exactly once.
// Dynamic LDS carve: [0,131072) fp32 scores | [131072,133120) bf16 q tile |
//                    [133120,134144) reduction scratch.
__global__ void attn_fused(const __bf16* __restrict__ qbf,
                           const __bf16* __restrict__ kbf,
                           const __bf16* __restrict__ vT,
                           const int* __restrict__ mask,
                           float* __restrict__ soft,
                           __bf16* __restrict__ attnbf) {
    extern __shared__ char smem_dyn[];
    float*  sc    = (float*)smem_dyn;
    __bf16* qtile = (__bf16*)(smem_dyn + 131072);
    float*  red   = (float*)(smem_dyn + 131072 + 2048);

    const int mtiles = Ssz >> 4;                 // 128
    const int mt = blockIdx.x % mtiles;
    const int bh = blockIdx.x / mtiles;
    const int b  = bh / Hh, h = bh % Hh;
    const int q0 = mt << 4;

    const int tid  = threadIdx.x;
    const int lane = tid & 31;
    const int w    = tid >> 5;
    const int lm   = lane & 15;
    const int kbaseA = (lane >> 4) << 3;
    const int khalfB = (lane >> 4) << 4;
    const int r0     = (lane >> 4) << 3;

    // stage q tile (16x64 bf16 = 1024 contiguous elements) into LDS
    const __bf16* qsrc = qbf + ((long)bh * Ssz + q0) * DHd;
#ifdef HAVE_TDM
    if (w == 0) {
        tdm_load_2d(131072u, qsrc, 1024u, 1u, 1024u);
        __builtin_amdgcn_s_wait_tensorcnt(0);
    }
#else
    if (tid < 128)
        *(v8bf*)&qtile[tid * 8] = *(const v8bf*)(qsrc + tid * 8);
#endif
    __syncthreads();

    // ---- phase 1: scores 16 x 2048; wave w owns columns [w*256, w*256+256) ----
    {
        const __bf16* kb = kbf + (long)bh * Ssz * DHd;
        v8f acc[16] = {};
        const __bf16* arow = qtile + lm * DHd;
#pragma unroll
        for (int k0 = 0; k0 < DHd; k0 += 32) {
            v16bf a = load_a_bf(arow + k0, kbaseA);
#pragma unroll
            for (int t = 0; t < 16; ++t) {
                const int col = (w << 8) + t * 16 + lm;
                v16bf bm = load_b_bf(kb + (long)col * DHd + k0, khalfB);
                acc[t] = wmma_bf16(a, bm, acc[t]);
            }
        }
        const int* mbase = mask + ((long)b * Ssz + q0) * Ssz;  // [B,1,S,S], bcast H
#pragma unroll
        for (int t = 0; t < 16; ++t) {
            const int col = (w << 8) + t * 16 + lm;
#pragma unroll
            for (int j = 0; j < 8; ++j) {
                const int row = r0 + j;
                float v = acc[t][j];
                if (mbase[(long)row * Ssz + col] != 0) v = -1e9f;
                sc[row * 2048 + col] = v;
            }
        }
    }
    __syncthreads();

    // ---- phase 2: row softmax; fp32 -> d_out (only HBM touch of soft_mat),
    //      bf16 copy kept in LDS (aliases low half of sc; row r's bf16 bytes
    //      overlap fp32 row r/2, consumed at step r/2 <= r; all reads complete
    //      before the post-reduction writes) ----
    __bf16* sbf = (__bf16*)sc;
    for (int row = 0; row < 16; ++row) {
        float vals[8];
        float mx = -3.4e38f;
#pragma unroll
        for (int j = 0; j < 8; ++j) {
            vals[j] = sc[row * 2048 + tid + j * 256];
            mx = fmaxf(mx, vals[j]);
        }
        red[tid] = mx; __syncthreads();
        for (int st = 128; st > 0; st >>= 1) { if (tid < st) red[tid] = fmaxf(red[tid], red[tid + st]); __syncthreads(); }
        mx = red[0]; __syncthreads();
        float sum = 0.0f;
#pragma unroll
        for (int j = 0; j < 8; ++j) { vals[j] = __expf(vals[j] - mx); sum += vals[j]; }
        red[tid] = sum; __syncthreads();
        for (int st = 128; st > 0; st >>= 1) { if (tid < st) red[tid] += red[tid + st]; __syncthreads(); }
        const float inv = 1.0f / red[0];
        float* gr = soft + ((long)bh * Ssz + q0 + row) * Ssz;
#pragma unroll
        for (int j = 0; j < 8; ++j) {
            const float v = vals[j] * inv;
            gr[tid + j * 256] = v;
            sbf[row * 2048 + tid + j * 256] = (__bf16)v;
        }
        __syncthreads();
    }

    // ---- phase 3: PV; K=2048 split across 8 waves, LDS reduction ----
    const __bf16* vb = vT + (long)bh * DHd * Ssz;
    v8f oacc[4] = {};
    const __bf16* arow2 = sbf + lm * 2048;
    for (int k0 = (w << 8); k0 < (w << 8) + 256; k0 += 32) {
        v16bf a = load_a_bf(arow2 + k0, kbaseA);
#pragma unroll
        for (int t = 0; t < 4; ++t) {
            v16bf bm = load_b_bf(vb + (long)(t * 16 + lm) * Ssz + k0, khalfB);
            oacc[t] = wmma_bf16(a, bm, oacc[t]);
        }
    }
    float* part = sc + 16 * 1024;   // 32 KB above the bf16 region: no overlap
#pragma unroll
    for (int t = 0; t < 4; ++t)
#pragma unroll
        for (int j = 0; j < 8; ++j)
            part[w * 1024 + t * 256 + j * 32 + lane] = oacc[t][j];
    __syncthreads();
    if (w < 4) {
        const int t = w;
#pragma unroll
        for (int j = 0; j < 8; ++j) {
            float s = 0.0f;
#pragma unroll
            for (int ww = 0; ww < 8; ++ww)
                s += part[ww * 1024 + t * 256 + j * 32 + lane];
            const int dh  = t * 16 + lm;
            const int row = q0 + r0 + j;
            attnbf[((long)(b * Ssz + row)) * Dsz + h * DHd + dh] = (__bf16)s;
        }
    }
}

// ---------------- host launch ----------------

extern "C" void kernel_launch(void* const* d_in, const int* in_sizes, int n_in,
                              void* d_out, int out_size, void* d_ws, size_t ws_size,
                              hipStream_t stream) {
    const float* Q  = (const float*)d_in[0];
    const float* K  = (const float*)d_in[1];
    const float* V  = (const float*)d_in[2];
    const int* mask = (const int*)d_in[3];
    const float* Wq = (const float*)d_in[4];  const float* bq = (const float*)d_in[5];
    const float* Wk = (const float*)d_in[6];  const float* bk = (const float*)d_in[7];
    const float* Wv = (const float*)d_in[8];  const float* bv = (const float*)d_in[9];
    const float* Wo = (const float*)d_in[10]; const float* bo = (const float*)d_in[11];

    float* out  = (float*)d_out;                      // [B,S,D]
    float* soft = out + (long)Bsz * Ssz * Dsz;        // [B,H,S,S]

    char* ws = (char*)d_ws;
    size_t off = 0;
    auto carve = [&](size_t bytes) -> void* {
        void* p = ws + off;
        off += (bytes + 255) & ~(size_t)255;
        return p;
    };
    const size_t bsd = (size_t)Bsz * Ssz * Dsz;       // 4M elems
    const size_t dd  = (size_t)Dsz * Dsz;             // 1M elems
    __bf16* Qbf  = (__bf16*)carve(bsd * 2);
    __bf16* Kbf  = (__bf16*)carve(bsd * 2);
    __bf16* Vbf  = (__bf16*)carve(bsd * 2);
    __bf16* WqT  = (__bf16*)carve(dd * 2);
    __bf16* WkT  = (__bf16*)carve(dd * 2);
    __bf16* WvT  = (__bf16*)carve(dd * 2);
    __bf16* WoT  = (__bf16*)carve(dd * 2);
    __bf16* qbf  = (__bf16*)carve(bsd * 2);           // [B,H,S,DH] pre-scaled
    __bf16* kbf  = (__bf16*)carve(bsd * 2);           // [B,H,S,DH]
    __bf16* vTbf = (__bf16*)carve(bsd * 2);           // [B,H,DH,S]
    __bf16* attn = (__bf16*)carve(bsd * 2);           // [B,S,D]

    const int M = Bsz * Ssz;                          // 4096
    const dim3 blk(256);

    cvt_f32_bf16<<<(int)(bsd / 256), blk, 0, stream>>>(Q, Qbf, (int)bsd);
    cvt_f32_bf16<<<(int)(bsd / 256), blk, 0, stream>>>(K, Kbf, (int)bsd);
    cvt_f32_bf16<<<(int)(bsd / 256), blk, 0, stream>>>(V, Vbf, (int)bsd);
    transpose_cvt<<<(int)(dd / 256), blk, 0, stream>>>(Wq, WqT, Dsz);
    transpose_cvt<<<(int)(dd / 256), blk, 0, stream>>>(Wk, WkT, Dsz);
    transpose_cvt<<<(int)(dd / 256), blk, 0, stream>>>(Wv, WvT, Dsz);
    transpose_cvt<<<(int)(dd / 256), blk, 0, stream>>>(Wo, WoT, Dsz);

    const int gemmBlocks = (M / 128) * (Dsz / 64);    // 512
    gemm_wmma<<<gemmBlocks, blk, 0, stream>>>(Qbf, Dsz, WqT, Dsz, bq, qbf,
                                              M, Dsz, Dsz, 0, 0.125f);
    gemm_wmma<<<gemmBlocks, blk, 0, stream>>>(Kbf, Dsz, WkT, Dsz, bk, kbf,
                                              M, Dsz, Dsz, 1, 1.0f);
    gemm_wmma<<<gemmBlocks, blk, 0, stream>>>(Vbf, Dsz, WvT, Dsz, bv, vTbf,
                                              M, Dsz, Dsz, 2, 1.0f);

    const size_t fusedLds = 16 * 2048 * 4 + 16 * DHd * 2 + 256 * 4;  // 134144 B
    attn_fused<<<Bsz * Hh * (Ssz / 16), blk, fusedLds, stream>>>(qbf, kbf, vTbf,
                                                                 mask, soft, attn);

    gemm_wmma<<<gemmBlocks, blk, 0, stream>>>(attn, Dsz, WoT, Dsz, bo, out,
                                              M, Dsz, Dsz, 3, 1.0f);
}
